// MultiHeadAttention_83820581749085
// MI455X (gfx1250) — matrix operand order repaired and checked by reading, not compiled, not run on hardware
//
#include <hip/hip_runtime.h>
#include <hip/hip_bf16.h>

// MI455X / gfx1250 multi-head attention forward.
// Pipeline: 3x WMMA-GEMM projections (f32 in -> f16 out, fp32 accum),
// flash-attention with v_wmma_f32_16x16x32_f16, final WMMA-GEMM to f32.
// Attention computes S^T = K @ Q^T and O^T = V^T @ P^T so each lane owns a
// query COLUMN: softmax reductions are in-register (+1 xor-16 shuffle),
// corr/l/m are per-lane scalars, and all LDS/global stores are b128.

typedef __attribute__((ext_vector_type(16))) _Float16 v16h;
typedef __attribute__((ext_vector_type(8)))  float    v8f;
typedef __attribute__((ext_vector_type(4)))  float    f32x4;
typedef __attribute__((ext_vector_type(2)))  unsigned int u32x2;

constexpr int DMODEL = 1024;
constexpr int NHEADS = 16;
constexpr int DK     = 64;
constexpr int SEQ    = 2048;
constexpr int BATCH  = 4;
constexpr int MROWS  = BATCH * SEQ;   // 8192

union Frag {
  v16h     v;
  f32x4    f4[2];
  _Float16 h[16];
};
union Frag8 { f32x4 f4; _Float16 h[8]; };
union Pack4 { u32x2 u; _Float16 h[4]; };

__device__ __forceinline__ v8f v8f_zero() {
  v8f z;
#pragma unroll
  for (int e = 0; e < 8; ++e) z[e] = 0.0f;
  return z;
}

__device__ __forceinline__ v8f wmma_f16(const Frag& a, const Frag& b, v8f c) {
  return __builtin_amdgcn_wmma_f32_16x16x32_f16(
      /*neg_a=*/false, a.v, /*neg_b=*/false, b.v,
      /*c_mod=*/(short)0, c, /*reuse_a=*/false, /*reuse_b=*/false);
}

__device__ __forceinline__ u32x2 cvt4(f32x4 f) {
  Pack4 pk;
  pk.h[0] = (_Float16)f[0]; pk.h[1] = (_Float16)f[1];
  pk.h[2] = (_Float16)f[2]; pk.h[3] = (_Float16)f[3];
  return pk.u;
}

// ---------------------------------------------------------------------------
// Tiled GEMM: C[M=8192][N=1024] = A[M][K=1024] @ W[K][N] + bias
// Block: 256 threads (8 waves), tile 128x128, K-step 32; wave tile 32x64.
// MODE 0: out f16 [M][N]      MODE 1: out f16 [B,H,dk,S]   MODE 2: out f32
// AHALF:  A matrix is f16 (attention output) vs f32 (x).
// ---------------------------------------------------------------------------
template <int MODE, bool AHALF>
__global__ __launch_bounds__(256)
void mha_gemm_kernel(const float* __restrict__ Af, const _Float16* __restrict__ Ah,
                     const float* __restrict__ Bw, const float* __restrict__ bias,
                     void* __restrict__ outp)
{
  constexpr int K = DMODEL, N = DMODEL;
  constexpr int LDS_STR = 40;                 // f16 units, padded
  __shared__ _Float16 sA[128 * LDS_STR];      // [m][k]
  __shared__ _Float16 sB[128 * LDS_STR];      // [n][k]  (W transposed in LDS)

  const int tid    = threadIdx.x;
  const int lane   = tid & 31;
  const int wave   = tid >> 5;
  const int m0     = (wave >> 1) * 32;
  const int n0     = (wave & 1) * 64;
  const int blockN = blockIdx.x * 128;
  const int blockM = blockIdx.y * 128;
  const int colIn  = lane & 15;
  const int half   = lane >> 4;

  v8f acc[2][4];
#pragma unroll
  for (int i = 0; i < 2; ++i)
#pragma unroll
    for (int j = 0; j < 4; ++j) acc[i][j] = v8f_zero();

  // register staging for software pipeline (raw data; convert at LDS store)
  f32x4 aStF[4];
  u32x2 aStH[4];
  f32x4 bStF[4];   // element e = W[k0 + kg*4 + e][n]  (column-wise)

  // ---- prefetch tile k0 = 0 ----
#pragma unroll
  for (int it = 0; it < 4; ++it) {
    int id = tid + it * 256;
    int row = id >> 3, c4 = (id & 7) * 4;
    if constexpr (AHALF)
      aStH[it] = *(const u32x2*)(Ah + (size_t)(blockM + row) * K + c4);
    else
      aStF[it] = *(const f32x4*)(Af + (size_t)(blockM + row) * K + c4);
  }
#pragma unroll
  for (int it = 0; it < 4; ++it) {
    int id = tid + it * 256;
    int n = id & 127, kg = id >> 7;
    const float* bp = Bw + (size_t)(kg * 4) * N + blockN + n;
#pragma unroll
    for (int e = 0; e < 4; ++e) bStF[it][e] = bp[(size_t)e * N];
  }

  for (int k0 = 0; k0 < K; k0 += 32) {
    // ---- commit staged tile to LDS (convert f32 -> f16 here) ----
#pragma unroll
    for (int it = 0; it < 4; ++it) {
      int id = tid + it * 256;
      int row = id >> 3, c4 = (id & 7) * 4;
      u32x2 u;
      if constexpr (AHALF) u = aStH[it];
      else                 u = cvt4(aStF[it]);
      *(u32x2*)&sA[row * LDS_STR + c4] = u;
    }
#pragma unroll
    for (int it = 0; it < 4; ++it) {
      int id = tid + it * 256;
      int n = id & 127, kg = id >> 7;
      u32x2 u = cvt4(bStF[it]);
      *(u32x2*)&sB[n * LDS_STR + kg * 4] = u;      // 4 contiguous k -> b64
    }
    __syncthreads();

    // ---- prefetch NEXT tile's global data (stays outstanding over WMMAs) --
    const int kn = k0 + 32;
    if (kn < K) {
#pragma unroll
      for (int it = 0; it < 4; ++it) {
        int id = tid + it * 256;
        int row = id >> 3, c4 = (id & 7) * 4;
        if constexpr (AHALF)
          aStH[it] = *(const u32x2*)(Ah + (size_t)(blockM + row) * K + kn + c4);
        else
          aStF[it] = *(const f32x4*)(Af + (size_t)(blockM + row) * K + kn + c4);
      }
#pragma unroll
      for (int it = 0; it < 4; ++it) {
        int id = tid + it * 256;
        int n = id & 127, kg = id >> 7;
        const float* bp = Bw + (size_t)(kn + kg * 4) * N + blockN + n;
#pragma unroll
        for (int e = 0; e < 4; ++e) bStF[it][e] = bp[(size_t)e * N];
      }
    }

    // ---- fragment loads (WMMA f16 lane layouts) + 8 WMMAs ----
    Frag aF[2], bF[4];
#pragma unroll
    for (int i = 0; i < 2; ++i) {
      int m = m0 + 16 * i + colIn;
      aF[i].f4[0] = *(const f32x4*)&sA[m * LDS_STR + half * 8];
      aF[i].f4[1] = *(const f32x4*)&sA[m * LDS_STR + half * 8 + 16];
    }
#pragma unroll
    for (int j = 0; j < 4; ++j) {
      int n = n0 + 16 * j + colIn;
      bF[j].f4[0] = *(const f32x4*)&sB[n * LDS_STR + half * 8];
      bF[j].f4[1] = *(const f32x4*)&sB[n * LDS_STR + half * 8 + 16];
    }
#pragma unroll
    for (int i = 0; i < 2; ++i)
#pragma unroll
      for (int j = 0; j < 4; ++j)
        acc[i][j] = wmma_f16(aF[i], bF[j], acc[i][j]);

    __syncthreads();   // LDS readers done before next commit
  }

  // ---- epilogue: bias add + store (C layout: lane=col, vgpr r = row) ----
#pragma unroll
  for (int j = 0; j < 4; ++j) {
    int nG = blockN + n0 + 16 * j + colIn;
    float bv = bias[nG];
#pragma unroll
    for (int i = 0; i < 2; ++i) {
#pragma unroll
      for (int r = 0; r < 8; ++r) {
        int mG = blockM + m0 + 16 * i + r + half * 8;
        float val = acc[i][j][r] + bv;
        if constexpr (MODE == 0) {
          ((_Float16*)outp)[(size_t)mG * N + nG] = (_Float16)val;
        } else if constexpr (MODE == 1) {
          int bb = mG >> 11;           // / SEQ
          int ss = mG & (SEQ - 1);
          int hh = nG >> 6;            // / DK
          int dd = nG & (DK - 1);
          ((_Float16*)outp)[(((size_t)bb * NHEADS + hh) * DK + dd) * SEQ + ss] =
              (_Float16)val;
        } else {
          ((float*)outp)[(size_t)mG * N + nG] = val;
        }
      }
    }
  }
}

// ---------------------------------------------------------------------------
// Flash attention (transposed): one wave per (b, h, 32-query tile).
// S^T = K_tile @ Q^T  -> lane owns a query column; per-lane softmax.
// O^T = V^T @ P^T     -> corr rescale is a per-lane scalar multiply.
// Q,K layout [B*S][DMODEL] f16; Vt layout [B,H,dk,S] f16.
// ---------------------------------------------------------------------------

// One 32-key tile. MASKED must be a literal so masking code folds away.
#define ATTN_TILE(KT, MASKED)                                                  \
  {                                                                            \
    const int kt_ = (KT);                                                      \
    /* K A-frags: 2 key groups x 2 dk-steps */                                 \
    Frag aK[2][2];                                                             \
    _Pragma("unroll") for (int gk = 0; gk < 2; ++gk) {                         \
      const _Float16* krow = Kb + (size_t)(kt_ + 16 * gk + m) * DMODEL;        \
      _Pragma("unroll") for (int s = 0; s < 2; ++s) {                          \
        aK[gk][s].f4[0] = *(const f32x4*)(krow + s * 32 + half * 8);           \
        aK[gk][s].f4[1] = *(const f32x4*)(krow + s * 32 + half * 8 + 16);      \
      }                                                                        \
    }                                                                          \
    /* S^T frags: rows = keys, cols = queries */                               \
    v8f st[2][2];                                                              \
    _Pragma("unroll") for (int gk = 0; gk < 2; ++gk)                           \
      _Pragma("unroll") for (int iq = 0; iq < 2; ++iq) {                       \
        v8f s_ = v8f_zero();                                                   \
        s_ = wmma_f16(aK[gk][0], bQ[iq][0], s_);                               \
        s_ = wmma_f16(aK[gk][1], bQ[iq][1], s_);                               \
        st[gk][iq] = s_;                                                       \
      }                                                                        \
    /* per-lane softmax over this lane's 16 keys; one xor-16 for the rest */   \
    _Pragma("unroll") for (int iq = 0; iq < 2; ++iq) {                         \
      float vv[2][8];                                                          \
      float vmax = NEG;                                                        \
      _Pragma("unroll") for (int gk = 0; gk < 2; ++gk)                         \
        _Pragma("unroll") for (int r = 0; r < 8; ++r) {                        \
          float v_ = st[gk][iq][r];                                            \
          if (MASKED) {                                                        \
            int key = kt_ + 16 * gk + r + 8 * half;                            \
            int q_  = qBase + 16 * iq + m;                                     \
            v_ = (key <= q_) ? v_ : NEG;                                       \
          }                                                                    \
          vv[gk][r] = v_;                                                      \
          vmax = fmaxf(vmax, v_);                                              \
        }                                                                      \
      vmax = fmaxf(vmax, __shfl_xor(vmax, 16, 32));                            \
      float mn   = fmaxf(mI[iq], vmax);                                        \
      float corr = __expf(mI[iq] - mn);                                        \
      float ps = 0.0f;                                                         \
      Frag8 pk_[2];                                                            \
      _Pragma("unroll") for (int gk = 0; gk < 2; ++gk)                         \
        _Pragma("unroll") for (int r = 0; r < 8; ++r) {                        \
          float p_ = __expf(vv[gk][r] - mn);                                   \
          ps += p_;                                                            \
          pk_[gk].h[r] = (_Float16)p_;                                         \
        }                                                                      \
      ps += __shfl_xor(ps, 16, 32);                                            \
      lI[iq] = lI[iq] * corr + ps;                                             \
      mI[iq] = mn;                                                             \
      /* P tile [query][key]: 8 contiguous keys -> one b128 store per gk */    \
      _Pragma("unroll") for (int gk = 0; gk < 2; ++gk)                         \
        *(f32x4*)&sP[wave][(16 * iq + m) * PSTR + 16 * gk + 8 * half] =        \
            pk_[gk].f4;                                                        \
      _Pragma("unroll") for (int jj = 0; jj < 4; ++jj)                         \
        _Pragma("unroll") for (int r = 0; r < 8; ++r)                          \
          accOT[jj][iq][r] *= corr;                                            \
    }                                                                          \
    /* P^T B-frags (from [query][key] tile) + V^T A-frags; O^T += V^T P^T */   \
    Frag bP[2];                                                                \
    _Pragma("unroll") for (int iq = 0; iq < 2; ++iq) {                         \
      bP[iq].f4[0] = *(const f32x4*)&sP[wave][(16*iq + m) * PSTR + half*8];    \
      bP[iq].f4[1] = *(const f32x4*)&sP[wave][(16*iq + m) * PSTR + half*8+16]; \
    }                                                                          \
    _Pragma("unroll") for (int jj = 0; jj < 4; ++jj) {                         \
      Frag aV;                                                                 \
      const _Float16* vrow = Vb + (size_t)(16 * jj + m) * SEQ + kt_;           \
      aV.f4[0] = *(const f32x4*)(vrow + half * 8);                             \
      aV.f4[1] = *(const f32x4*)(vrow + half * 8 + 16);                        \
      _Pragma("unroll") for (int iq = 0; iq < 2; ++iq)                         \
        accOT[jj][iq] = wmma_f16(aV, bP[iq], accOT[jj][iq]);                   \
    }                                                                          \
  }

__global__ __launch_bounds__(128)
void mha_attn_kernel(const _Float16* __restrict__ Q, const _Float16* __restrict__ Kh,
                     const _Float16* __restrict__ Vt, _Float16* __restrict__ O)
{
  constexpr int PSTR = 40;                  // padded P-tile stride (f16)
  __shared__ _Float16 sP[4][32 * PSTR];     // one 32x32 P tile per wave

  const int lane  = threadIdx.x & 31;
  const int wave  = threadIdx.x >> 5;
  const int m     = lane & 15;
  const int half  = lane >> 4;
  const int bh    = blockIdx.y;
  const int b     = bh >> 4;
  const int h     = bh & 15;
  const int qBase = (blockIdx.x * 4 + wave) * 32;
  const float NEG = -__builtin_inff();

  // --- Q^T B-frags: 2 query blocks x 2 dk-steps, pre-scaled by 1/sqrt(dk) ---
  Frag bQ[2][2];
#pragma unroll
  for (int iq = 0; iq < 2; ++iq) {
    const _Float16* qrow =
        Q + ((size_t)(b * SEQ + qBase + 16 * iq + m)) * DMODEL + h * DK;
#pragma unroll
    for (int s = 0; s < 2; ++s) {
      bQ[iq][s].f4[0] = *(const f32x4*)(qrow + s * 32 + half * 8);
      bQ[iq][s].f4[1] = *(const f32x4*)(qrow + s * 32 + half * 8 + 16);
#pragma unroll
      for (int e = 0; e < 16; ++e)
        bQ[iq][s].h[e] = bQ[iq][s].h[e] * (_Float16)0.125f;
    }
  }

  const _Float16* Kb = Kh + (size_t)b * SEQ * DMODEL + h * DK;
  const _Float16* Vb = Vt + ((size_t)(b * NHEADS + h)) * DK * SEQ;

  float mI[2], lI[2];
  v8f accOT[4][2];            // [dk block][query block]; rows = dk, cols = q
  mI[0] = mI[1] = NEG;
  lI[0] = lI[1] = 0.0f;
#pragma unroll
  for (int jj = 0; jj < 4; ++jj)
#pragma unroll
    for (int iq = 0; iq < 2; ++iq) accOT[jj][iq] = v8f_zero();

  // ---- full (never-masked) key tiles: kt < qBase ----
  const int fullT = qBase >> 5;
  for (int t = 0; t < fullT; ++t) {
    ATTN_TILE(t * 32, false);
  }
  // ---- diagonal tile (the only one that can mask): kt == qBase ----
  ATTN_TILE(qBase, true);

  // ---- normalize; store O^T frag columns as 8 contiguous dk -> b128 ----
#pragma unroll
  for (int iq = 0; iq < 2; ++iq) {
    float invL = 1.0f / lI[iq];
    _Float16* Op =
        O + ((size_t)(b * SEQ + qBase + 16 * iq + m)) * DMODEL + h * DK;
#pragma unroll
    for (int jj = 0; jj < 4; ++jj) {
      Frag8 o;
#pragma unroll
      for (int r = 0; r < 8; ++r)
        o.h[r] = (_Float16)(accOT[jj][iq][r] * invL);
      *(f32x4*)(Op + 16 * jj + 8 * half) = o.f4;
    }
  }
}
#undef ATTN_TILE

// ---------------------------------------------------------------------------
extern "C" void kernel_launch(void* const* d_in, const int* in_sizes, int n_in,
                              void* d_out, int out_size, void* d_ws, size_t ws_size,
                              hipStream_t stream) {
  (void)in_sizes; (void)n_in; (void)out_size; (void)ws_size;
  const float* x  = (const float*)d_in[0];
  const float* Wq = (const float*)d_in[1];
  const float* bq = (const float*)d_in[2];
  const float* Wk = (const float*)d_in[3];
  const float* bk = (const float*)d_in[4];
  const float* Wv = (const float*)d_in[5];
  const float* bv = (const float*)d_in[6];
  const float* Wo = (const float*)d_in[7];
  const float* bo = (const float*)d_in[8];

  char* ws = (char*)d_ws;
  const size_t SZ = (size_t)MROWS * DMODEL * sizeof(_Float16);   // 16 MB each
  _Float16* Qh = (_Float16*)(ws + 0 * SZ);
  _Float16* Kh = (_Float16*)(ws + 1 * SZ);
  _Float16* Vt = (_Float16*)(ws + 2 * SZ);
  _Float16* Oh = (_Float16*)(ws + 3 * SZ);

  dim3 gBlk(256), gGrd(DMODEL / 128, MROWS / 128);               // (8, 64)
  mha_gemm_kernel<0, false><<<gGrd, gBlk, 0, stream>>>(x, nullptr, Wq, bq, Qh);
  mha_gemm_kernel<0, false><<<gGrd, gBlk, 0, stream>>>(x, nullptr, Wk, bk, Kh);
  mha_gemm_kernel<1, false><<<gGrd, gBlk, 0, stream>>>(x, nullptr, Wv, bv, Vt);

  dim3 aBlk(128), aGrd(SEQ / 128, BATCH * NHEADS);               // (16, 64)
  mha_attn_kernel<<<aGrd, aBlk, 0, stream>>>(Qh, Kh, Vt, Oh);

  mha_gemm_kernel<2, true><<<gGrd, gBlk, 0, stream>>>(nullptr, Oh, Wo, bo, d_out);
}
